// HandCodedAttentionLayer_28862180229774
// MI455X (gfx1250) — matrix operand order repaired and checked by reading, not compiled, not run on hardware
//
#include <hip/hip_runtime.h>
#include <hip/hip_bf16.h>

typedef __bf16 bf16;
typedef __attribute__((ext_vector_type(8)))  __bf16 v8bf;
typedef __attribute__((ext_vector_type(16))) __bf16 v16bf;
typedef __attribute__((ext_vector_type(8)))  float  v8f;

#define B_ 4
#define S_ 2048
#define E_ 1024
#define H_ 16
#define D_ 64
#define M_ (B_*S_)

// ds_swizzle_b32 group-of-32 XOR pattern: offset = {xor[14:10], or[9:5], and[4:0]}
#define SWZ_XOR(x, m) \
  __int_as_float(__builtin_amdgcn_ds_swizzle(__float_as_int(x), ((m) << 10) | 0x1f))

// ---------------------------------------------------------------------------
// CDNA5 async memory->LDS copy (GLOBAL_LOAD_ASYNC_TO_LDS_B128, ASYNCcnt)
// ---------------------------------------------------------------------------
__device__ __forceinline__ void async_copy_b128(void* lds_ptr, const void* gptr) {
  unsigned off = (unsigned)(unsigned long long)(uintptr_t)lds_ptr;
  asm volatile("global_load_async_to_lds_b128 %0, %1, off"
               :: "v"(off), "v"(gptr) : "memory");
}
__device__ __forceinline__ void wait_async0() {
  asm volatile("s_wait_asynccnt 0x0" ::: "memory");
}

// ---------------------------------------------------------------------------
// WMMA fragment helpers (CDNA5 wave32 layouts, CDNA5 ISA §7.12.2)
// ---------------------------------------------------------------------------
__device__ __forceinline__ v16bf frag_a_ld(const bf16* base, int stride, int lane) {
  const int m = lane & 15, g = lane >> 4;
  const bf16* p = base + m * stride + 8 * g;
  v8bf lo = *(const v8bf*)(p);
  v8bf hi = *(const v8bf*)(p + 16);
  return __builtin_shufflevector(lo, hi, 0,1,2,3,4,5,6,7,8,9,10,11,12,13,14,15);
}

__device__ __forceinline__ v16bf frag_b_ld(const bf16* base, int stride, int lane) {
  const int n = lane & 15, g = lane >> 4;
  const bf16* p = base + n * stride + 8 * g;
  v8bf lo = *(const v8bf*)(p);
  v8bf hi = *(const v8bf*)(p + 16);
  return __builtin_shufflevector(lo, hi, 0,1,2,3,4,5,6,7,8,9,10,11,12,13,14,15);
}

__device__ __forceinline__ v8f wmma_bf16(v16bf a, v16bf b, v8f c) {
  return __builtin_amdgcn_wmma_f32_16x16x32_bf16(false, a, false, b, (short)0, c,
                                                 false, false);
}

// ---------------------------------------------------------------------------
// fp32 -> bf16 conversion
// ---------------------------------------------------------------------------
__global__ void cvt_f32_bf16(const float* __restrict__ in, bf16* __restrict__ out,
                             int n) {
  int i = (blockIdx.x * blockDim.x + threadIdx.x) * 4;
  if (i < n) {
    float4 v = *(const float4*)(in + i);
    out[i + 0] = (bf16)v.x;
    out[i + 1] = (bf16)v.y;
    out[i + 2] = (bf16)v.z;
    out[i + 3] = (bf16)v.w;
  }
}

// ---------------------------------------------------------------------------
// GEMM: Y[M,N] = A[M,K] @ W[N,K]^T + bias, async double-buffered LDS pipeline
// Block tile 128x128, BK=32, 256 threads = 8 waves (2Mx4N), 8 WMMA/wave/step.
// ---------------------------------------------------------------------------
template <bool OUT_BF16>
__global__ __launch_bounds__(256)
void gemm_wmma_nt(const bf16* __restrict__ A, const bf16* __restrict__ W,
                  const float* __restrict__ bias, bf16* __restrict__ Yb,
                  float* __restrict__ Yf, int Mdim, int Ndim, int Kdim) {
  __shared__ __align__(16) bf16 As[2][128 * 32];
  __shared__ __align__(16) bf16 Bs[2][128 * 32];

  const int tid  = threadIdx.x;
  const int lane = tid & 31;
  const int wid  = tid >> 5;
  const int wm   = wid & 1;   // 0..1
  const int wn   = wid >> 1;  // 0..3
  const int bm   = blockIdx.y * 128;
  const int bn   = blockIdx.x * 128;

  v8f acc[4][2];
#pragma unroll
  for (int mi = 0; mi < 4; mi++)
#pragma unroll
    for (int ni = 0; ni < 2; ni++)
#pragma unroll
      for (int r = 0; r < 8; r++) acc[mi][ni][r] = 0.0f;

  const int ldr = tid >> 1;        // 0..127
  const int ldc = (tid & 1) * 16;  // 0 or 16
  const bf16* gA = A + (size_t)(bm + ldr) * Kdim + ldc;
  const bf16* gB = W + (size_t)(bn + ldr) * Kdim + ldc;

  auto issue = [&](int kofs, int buf) {
    async_copy_b128(&As[buf][ldr * 32 + ldc],     gA + kofs);
    async_copy_b128(&As[buf][ldr * 32 + ldc + 8], gA + kofs + 8);
    async_copy_b128(&Bs[buf][ldr * 32 + ldc],     gB + kofs);
    async_copy_b128(&Bs[buf][ldr * 32 + ldc + 8], gB + kofs + 8);
  };

  issue(0, 0);
  wait_async0();
  __syncthreads();

  const int nk = Kdim >> 5;
  for (int i = 0; i < nk; i++) {
    if (i + 1 < nk) issue((i + 1) * 32, (i + 1) & 1);  // overlap with compute

    const bf16* as = As[i & 1];
    const bf16* bs = Bs[i & 1];
    v16bf af[4], bfr[2];
#pragma unroll
    for (int mi = 0; mi < 4; mi++)
      af[mi] = frag_a_ld(as + (wm * 64 + mi * 16) * 32, 32, lane);
#pragma unroll
    for (int ni = 0; ni < 2; ni++)
      bfr[ni] = frag_b_ld(bs + (wn * 32 + ni * 16) * 32, 32, lane);
#pragma unroll
    for (int mi = 0; mi < 4; mi++)
#pragma unroll
      for (int ni = 0; ni < 2; ni++)
        acc[mi][ni] = wmma_bf16(af[mi], bfr[ni], acc[mi][ni]);

    wait_async0();   // next tile landed in LDS
    __syncthreads();
  }

  const int g  = lane >> 4;
  const int cL = lane & 15;
#pragma unroll
  for (int ni = 0; ni < 2; ni++) {
    const int col = bn + wn * 32 + ni * 16 + cL;
    const float bv = bias[col];
#pragma unroll
    for (int mi = 0; mi < 4; mi++) {
#pragma unroll
      for (int r = 0; r < 8; r++) {
        const int row = bm + wm * 64 + mi * 16 + r + 8 * g;
        const float v = acc[mi][ni][r] + bv;
        if (OUT_BF16)
          Yb[(size_t)row * Ndim + col] = (bf16)v;
        else
          Yf[(size_t)row * Ndim + col] = v;
      }
    }
  }
}

// ---------------------------------------------------------------------------
// Flash attention: grid (S/128, B*H), 256 threads = 8 waves.
// K chunk staged via async-to-LDS; V transposed through registers.
// Per chunk per wave: 4 WMMA (scores) + online softmax + 4 WMMA (P@V).
// ---------------------------------------------------------------------------
__global__ __launch_bounds__(256)
void attn_flash(const bf16* __restrict__ Q, const bf16* __restrict__ K,
                const bf16* __restrict__ V, bf16* __restrict__ O) {
  __shared__ __align__(16) bf16 Ks[32 * 64];      // [key][d]
  __shared__ __align__(16) bf16 Vt[64 * 32];      // [d][key] (transposed)
  __shared__ __align__(16) bf16 Ps[8 * 16 * 32];  // per-wave P staging

  const int tid  = threadIdx.x;
  const int lane = tid & 31;
  const int w    = tid >> 5;
  const int g    = lane >> 4;
  const int cL   = lane & 15;

  const int bh = blockIdx.y;
  const int bb = bh >> 4;  // H_ == 16
  const int hh = bh & 15;
  const size_t head = (size_t)bb * S_ * E_ + (size_t)hh * D_;
  const int q0 = blockIdx.x * 128 + w * 16;

  const bf16* qh = Q + head;
  const bf16* kh = K + head;
  const bf16* vh = V + head;

  const v16bf aQ0 = frag_a_ld(qh + (size_t)q0 * E_, E_, lane);
  const v16bf aQ1 = frag_a_ld(qh + (size_t)q0 * E_ + 32, E_, lane);

  v8f oacc[4];
  float mr[8], lr[8];
#pragma unroll
  for (int dt = 0; dt < 4; dt++)
#pragma unroll
    for (int r = 0; r < 8; r++) oacc[dt][r] = 0.0f;
#pragma unroll
  for (int r = 0; r < 8; r++) { mr[r] = -1.0e30f; lr[r] = 0.0f; }

  bf16* pw = Ps + w * (16 * 32);
  const int ldk = tid >> 3;       // key row 0..31
  const int ldd = (tid & 7) * 8;  // d offset

  const float inv_scale = 0.125f;  // 1/sqrt(D), D=64

  for (int j0 = 0; j0 < S_; j0 += 32) {
    __syncthreads();  // previous chunk's compute done -> safe to overwrite
    async_copy_b128(&Ks[ldk * 64 + ldd], kh + (size_t)(j0 + ldk) * E_ + ldd);
    {
      v8bf vv = *(const v8bf*)(vh + (size_t)(j0 + ldk) * E_ + ldd);
#pragma unroll
      for (int i = 0; i < 8; i++) Vt[(ldd + i) * 32 + ldk] = vv[i];
    }
    wait_async0();
    __syncthreads();

    // scores: two 16-key column tiles, K-dim = D = 64 (2 chained WMMAs each)
    v8f s0, s1;
#pragma unroll
    for (int r = 0; r < 8; r++) { s0[r] = 0.0f; s1[r] = 0.0f; }
    {
      v16bf b00 = frag_b_ld(Ks, 64, lane);
      v16bf b01 = frag_b_ld(Ks + 32, 64, lane);
      s0 = wmma_bf16(aQ0, b00, s0);
      s0 = wmma_bf16(aQ1, b01, s0);
      v16bf b10 = frag_b_ld(Ks + 16 * 64, 64, lane);
      v16bf b11 = frag_b_ld(Ks + 16 * 64 + 32, 64, lane);
      s1 = wmma_bf16(aQ0, b10, s1);
      s1 = wmma_bf16(aQ1, b11, s1);
    }

    // online softmax; rows at r+8g, 16 cols across half-wave lanes.
    // Cross-lane reductions via ds_swizzle_b32 XOR patterns (SWAPX1/2/4/8).
    float sc[8];
#pragma unroll
    for (int r = 0; r < 8; r++) {
      float x0 = s0[r] * inv_scale;
      float x1 = s1[r] * inv_scale;
      float vmax = fmaxf(x0, x1);
      vmax = fmaxf(vmax, SWZ_XOR(vmax, 1));
      vmax = fmaxf(vmax, SWZ_XOR(vmax, 2));
      vmax = fmaxf(vmax, SWZ_XOR(vmax, 4));
      vmax = fmaxf(vmax, SWZ_XOR(vmax, 8));
      const float mn = fmaxf(mr[r], vmax);
      sc[r] = __expf(mr[r] - mn);
      mr[r] = mn;
      const float p0 = __expf(x0 - mn);
      const float p1 = __expf(x1 - mn);
      s0[r] = p0;
      s1[r] = p1;
      float ps = p0 + p1;
      ps += SWZ_XOR(ps, 1);
      ps += SWZ_XOR(ps, 2);
      ps += SWZ_XOR(ps, 4);
      ps += SWZ_XOR(ps, 8);
      lr[r] = lr[r] * sc[r] + ps;
    }
#pragma unroll
    for (int dt = 0; dt < 4; dt++)
#pragma unroll
      for (int r = 0; r < 8; r++) oacc[dt][r] *= sc[r];

    // stage P (C-layout -> LDS [m][kk]) then reload as A-fragment
#pragma unroll
    for (int r = 0; r < 8; r++) {
      pw[(r + 8 * g) * 32 + cL]      = (bf16)s0[r];
      pw[(r + 8 * g) * 32 + 16 + cL] = (bf16)s1[r];
    }
    asm volatile("s_wait_dscnt 0x0" ::: "memory");
    const v16bf aP = frag_a_ld(pw, 32, lane);

#pragma unroll
    for (int dt = 0; dt < 4; dt++) {
      v16bf bv = frag_b_ld(Vt + dt * 16 * 32, 32, lane);
      oacc[dt] = wmma_bf16(aP, bv, oacc[dt]);
    }
  }

  // normalize and store context (bf16, [B,S,E] layout with head offset)
#pragma unroll
  for (int r = 0; r < 8; r++) lr[r] = 1.0f / lr[r];
  bf16* oh = O + head;
#pragma unroll
  for (int dt = 0; dt < 4; dt++) {
#pragma unroll
    for (int r = 0; r < 8; r++) {
      const int row = q0 + r + 8 * g;
      oh[(size_t)row * E_ + dt * 16 + cL] = (bf16)(oacc[dt][r] * lr[r]);
    }
  }
}

// ---------------------------------------------------------------------------
// Launch
// ---------------------------------------------------------------------------
extern "C" void kernel_launch(void* const* d_in, const int* in_sizes, int n_in,
                              void* d_out, int out_size, void* d_ws,
                              size_t ws_size, hipStream_t stream) {
  const float* x  = (const float*)d_in[0];
  const float* Wq = (const float*)d_in[1];
  const float* bq = (const float*)d_in[2];
  const float* Wk = (const float*)d_in[3];
  const float* bk = (const float*)d_in[4];
  const float* Wv = (const float*)d_in[5];
  const float* bv = (const float*)d_in[6];
  const float* Wo = (const float*)d_in[7];
  const float* bo = (const float*)d_in[8];
  float* out = (float*)d_out;

  char* ws = (char*)d_ws;
  const size_t MB = 1024ull * 1024ull;
  bf16* xb  = (bf16*)(ws + 0 * MB);
  bf16* wqb = (bf16*)(ws + 16 * MB);
  bf16* wkb = (bf16*)(ws + 18 * MB);
  bf16* wvb = (bf16*)(ws + 20 * MB);
  bf16* wob = (bf16*)(ws + 22 * MB);
  bf16* qb  = (bf16*)(ws + 24 * MB);
  bf16* kb  = (bf16*)(ws + 40 * MB);
  bf16* vb  = (bf16*)(ws + 56 * MB);
  bf16* cb  = (bf16*)(ws + 72 * MB);

  const int nx = M_ * E_;
  const int nw = E_ * E_;
  cvt_f32_bf16<<<nx / 4 / 256, 256, 0, stream>>>(x, xb, nx);
  cvt_f32_bf16<<<nw / 4 / 256, 256, 0, stream>>>(Wq, wqb, nw);
  cvt_f32_bf16<<<nw / 4 / 256, 256, 0, stream>>>(Wk, wkb, nw);
  cvt_f32_bf16<<<nw / 4 / 256, 256, 0, stream>>>(Wv, wvb, nw);
  cvt_f32_bf16<<<nw / 4 / 256, 256, 0, stream>>>(Wo, wob, nw);

  dim3 gg(E_ / 128, M_ / 128);  // (8, 64)
  gemm_wmma_nt<true><<<gg, dim3(256), 0, stream>>>(xb, wqb, bq, qb, nullptr,
                                                   M_, E_, E_);
  gemm_wmma_nt<true><<<gg, dim3(256), 0, stream>>>(xb, wkb, bk, kb, nullptr,
                                                   M_, E_, E_);
  gemm_wmma_nt<true><<<gg, dim3(256), 0, stream>>>(xb, wvb, bv, vb, nullptr,
                                                   M_, E_, E_);

  dim3 ga(S_ / 128, B_ * H_);  // (16, 64)
  attn_flash<<<ga, dim3(256), 0, stream>>>(qb, kb, vb, cb);

  gemm_wmma_nt<false><<<gg, dim3(256), 0, stream>>>(cb, wob, bo, nullptr, out,
                                                    M_, E_, E_);
}
